// Memory_41291815584279
// MI455X (gfx1250) — matrix operand order repaired and checked by reading, not compiled, not run on hardware
//
#include <hip/hip_runtime.h>
#include <stdint.h>

// Problem dims
#define NO      4
#define CENTERS 2048
#define CDIM    256
#define VDDIM   512
#define HWDIM   2304   // 48*48

typedef __attribute__((ext_vector_type(8)))  float        v8f;
typedef __attribute__((ext_vector_type(16))) _Float16     v16h;
typedef __attribute__((ext_vector_type(4)))  unsigned int v4u;

union HFrag { v4u u[2]; v16h v; };

__device__ __forceinline__ unsigned short f2h(float f) {
  _Float16 h = (_Float16)f;            // v_cvt_f16_f32, RNE
  unsigned short u;
  __builtin_memcpy(&u, &h, 2);
  return u;
}
__device__ __forceinline__ unsigned int packh(float a, float b) {
  return (unsigned int)f2h(a) | ((unsigned int)f2h(b) << 16);
}

// ---------------- pre-pass kernels ----------------

__global__ void cvt_f16_kernel(const float* __restrict__ src,
                               unsigned short* __restrict__ dst, int n2) {
  int i = blockIdx.x * blockDim.x + threadIdx.x;
  if (i < n2) {
    int j = i * 2;
    unsigned int p = packh(src[j], src[j + 1]);
    *(unsigned int*)&dst[j] = p;
  }
}

// m_out [n][k][vd] (f32) -> Vt [n][vd][k] (f16), LDS-tiled 32x32
__global__ void transpose_vt_kernel(const float* __restrict__ src,
                                    unsigned short* __restrict__ dst) {
  __shared__ float tile[32][33];
  const int k0 = blockIdx.x * 32;
  const int v0 = blockIdx.y * 32;
  const int n  = blockIdx.z;
  const float* s = src + (size_t)n * CENTERS * VDDIM;
  unsigned short* d = dst + (size_t)n * VDDIM * CENTERS;
  const int tx = threadIdx.x, ty = threadIdx.y;
  for (int r = ty; r < 32; r += 8)
    tile[r][tx] = s[(size_t)(k0 + r) * VDDIM + v0 + tx];
  __syncthreads();
  for (int r = ty; r < 32; r += 8)
    d[(size_t)(v0 + r) * CENTERS + k0 + tx] = f2h(tile[tx][r]);
}

// q_out -> out channels [VD, 2*VD)
__global__ void copy_qout_kernel(const float* __restrict__ qo,
                                 float* __restrict__ out) {
  int i = blockIdx.x * blockDim.x + threadIdx.x;
  const int per = VDDIM * HWDIM;
  if (i < NO * per) {
    int n = i / per;
    int r = i - n * per;
    out[(size_t)n * (2 * VDDIM) * HWDIM + (size_t)VDDIM * HWDIM + r] = qo[i];
  }
}

// ---------------- fused flash-attention kernel ----------------

#define QS_STRIDE 264   // 256 + 8 f16 pad (bank spread)
#define PS_STRIDE 136   // 128 + 8 f16 pad

__global__ __launch_bounds__(256) void flash_attn_kernel(
    const unsigned short* __restrict__ Kb,    // m_in f16    [n][2048][256]
    const unsigned short* __restrict__ pKb,   // p_m_in f16
    const float*          __restrict__ Qf,    // q_in f32    [n][256][2304]
    const float*          __restrict__ pQf,   // p_q_in f32
    const unsigned short* __restrict__ Vt,    // m_out^T f16 [n][512][2048]
    float*                __restrict__ out)   // [n][1024][2304]
{
  __shared__ __align__(16) unsigned short Qs[32 * QS_STRIDE];
  __shared__ __align__(16) unsigned short Ps[32 * PS_STRIDE];
  __shared__ float redA[8 * 32];
  __shared__ float redB[8 * 32];

  const int tid  = threadIdx.x;
  const int w    = tid >> 5;      // wave 0..7, owns vd slice [64w, 64w+64)
  const int lane = tid & 31;
  const int ln   = lane & 15;
  const int lh   = lane >> 4;

  const int q0 = blockIdx.x * 32; // 32 queries per workgroup
  const int n  = blockIdx.y;

  float* outp = out + (size_t)n * (2 * VDDIM) * HWDIM;

  for (int br = 0; br < 2; ++br) {
    const unsigned short* Kp = (br ? pKb : Kb) + (size_t)n * CENTERS * CDIM;
    const float*          Qp = (br ? pQf : Qf) + (size_t)n * CDIM * HWDIM;

    __syncthreads();
    // Stage Q tile (32 q x 256 c) into LDS as f16, layout Qs[q][c]
    {
      const int q  = tid & 31;
      const int cw = tid >> 5;
      for (int it = 0; it < 32; ++it) {
        int c = it * 8 + cw;
        Qs[q * QS_STRIDE + c] = f2h(Qp[(size_t)c * HWDIM + q0 + q]);
      }
    }
    __syncthreads();

    float m_run0 = -1e30f, m_run1 = -1e30f;
    float l_run0 = 0.f,    l_run1 = 0.f;
    v8f acc[4][2];
    #pragma unroll
    for (int vt = 0; vt < 4; ++vt)
      #pragma unroll
      for (int i = 0; i < 8; ++i) { acc[vt][0][i] = 0.f; acc[vt][1][i] = 0.f; }

    const unsigned short* Arow0 = Kp + (size_t)(w * 16 + ln) * CDIM + lh * 8;
    const unsigned short* Vrow  = Vt + (size_t)n * VDDIM * CENTERS
                                     + (size_t)(w * 64 + ln) * CENTERS + lh * 8;

    for (int k0 = 0; k0 < CENTERS; k0 += 128) {
      const unsigned short* Arow = Arow0 + (size_t)k0 * CDIM;
      if (k0 + 128 < CENTERS) __builtin_prefetch(Arow + 128 * CDIM, 0, 1);

      // ---- GEMM1: S[16k x 16q] x2 q-tiles, K=256 in 8 chunks of 32 ----
      v8f s0, s1;
      #pragma unroll
      for (int i = 0; i < 8; ++i) { s0[i] = 0.f; s1[i] = 0.f; }

      #pragma unroll
      for (int kc = 0; kc < 8; ++kc) {
        HFrag a, b0, b1;
        a.u[0] = *(const v4u*)(Arow + kc * 32);
        a.u[1] = *(const v4u*)(Arow + kc * 32 + 16);
        const unsigned short* B0 = &Qs[ln * QS_STRIDE + kc * 32 + lh * 8];
        const unsigned short* B1 = B0 + 16 * QS_STRIDE;
        b0.u[0] = *(const v4u*)(B0);
        b0.u[1] = *(const v4u*)(B0 + 16);
        b1.u[0] = *(const v4u*)(B1);
        b1.u[1] = *(const v4u*)(B1 + 16);
        s0 = __builtin_amdgcn_wmma_f32_16x16x32_f16(false, a.v, false, b0.v,
                                                    (short)0, s0, false, false);
        s1 = __builtin_amdgcn_wmma_f32_16x16x32_f16(false, a.v, false, b1.v,
                                                    (short)0, s1, false, false);
      }

      // ---- online softmax: column (per-query) max over 128 keys ----
      float cm0 = -1e30f, cm1 = -1e30f;
      #pragma unroll
      for (int i = 0; i < 8; ++i) {
        s0[i] *= 0.0625f; s1[i] *= 0.0625f;   // 1/sqrt(256)
        cm0 = fmaxf(cm0, s0[i]); cm1 = fmaxf(cm1, s1[i]);
      }
      cm0 = fmaxf(cm0, __shfl_xor(cm0, 16, 32));
      cm1 = fmaxf(cm1, __shfl_xor(cm1, 16, 32));
      if (lane < 16) { redA[w * 32 + ln] = cm0; redA[w * 32 + 16 + ln] = cm1; }
      __syncthreads();
      float mb0 = -1e30f, mb1 = -1e30f;
      #pragma unroll
      for (int ww = 0; ww < 8; ++ww) {
        mb0 = fmaxf(mb0, redA[ww * 32 + ln]);
        mb1 = fmaxf(mb1, redA[ww * 32 + 16 + ln]);
      }
      float mn0 = fmaxf(m_run0, mb0), mn1 = fmaxf(m_run1, mb1);
      float c0 = __expf(m_run0 - mn0), c1 = __expf(m_run1 - mn1);
      m_run0 = mn0; m_run1 = mn1;

      // ---- P = exp(S - m), pack f16 into Ps[q][k] + row sums ----
      float ls0 = 0.f, ls1 = 0.f;
      v4u pk0, pk1;
      #pragma unroll
      for (int j = 0; j < 4; ++j) {
        float pa = __expf(s0[2 * j] - mn0), pb = __expf(s0[2 * j + 1] - mn0);
        ls0 += pa + pb; pk0[j] = packh(pa, pb);
        float pc = __expf(s1[2 * j] - mn1), pd = __expf(s1[2 * j + 1] - mn1);
        ls1 += pc + pd; pk1[j] = packh(pc, pd);
      }
      *(v4u*)&Ps[ln * PS_STRIDE + w * 16 + lh * 8]        = pk0;
      *(v4u*)&Ps[(16 + ln) * PS_STRIDE + w * 16 + lh * 8] = pk1;

      ls0 += __shfl_xor(ls0, 16, 32);
      ls1 += __shfl_xor(ls1, 16, 32);
      if (lane < 16) { redB[w * 32 + ln] = ls0; redB[w * 32 + 16 + ln] = ls1; }
      __syncthreads();   // Ps visible + redB ready
      float la0 = 0.f, la1 = 0.f;
      #pragma unroll
      for (int ww = 0; ww < 8; ++ww) {
        la0 += redB[ww * 32 + ln];
        la1 += redB[ww * 32 + 16 + ln];
      }
      l_run0 = l_run0 * c0 + la0;
      l_run1 = l_run1 * c1 + la1;

      // rescale accumulators by exp(m_old - m_new)
      #pragma unroll
      for (int vt = 0; vt < 4; ++vt)
        #pragma unroll
        for (int i = 0; i < 8; ++i) { acc[vt][0][i] *= c0; acc[vt][1][i] *= c1; }

      // ---- GEMM2: O[64vd x 16q] += Vt[64vd x 128k] * P[128k x 16q] ----
      const unsigned short* Vk = Vrow + k0;
      if (k0 + 128 < CENTERS) __builtin_prefetch(Vk + 128, 0, 1);
      #pragma unroll
      for (int ks = 0; ks < 4; ++ks) {
        HFrag pb0, pb1;
        const unsigned short* P0 = &Ps[ln * PS_STRIDE + ks * 32 + lh * 8];
        const unsigned short* P1 = P0 + 16 * PS_STRIDE;
        pb0.u[0] = *(const v4u*)(P0);
        pb0.u[1] = *(const v4u*)(P0 + 16);
        pb1.u[0] = *(const v4u*)(P1);
        pb1.u[1] = *(const v4u*)(P1 + 16);
        #pragma unroll
        for (int vt = 0; vt < 4; ++vt) {
          HFrag av;
          const unsigned short* Ap = Vk + (size_t)vt * 16 * CENTERS + ks * 32;
          av.u[0] = *(const v4u*)(Ap);
          av.u[1] = *(const v4u*)(Ap + 16);
          acc[vt][0] = __builtin_amdgcn_wmma_f32_16x16x32_f16(
              false, av.v, false, pb0.v, (short)0, acc[vt][0], false, false);
          acc[vt][1] = __builtin_amdgcn_wmma_f32_16x16x32_f16(
              false, av.v, false, pb1.v, (short)0, acc[vt][1], false, false);
        }
      }
      __syncthreads();   // protect Ps/red before next iteration
    }

    // ---- finalize: normalize by row sums; br0 stores, br1 accumulates ----
    float r0 = 1.f / l_run0, r1 = 1.f / l_run1;
    #pragma unroll
    for (int vt = 0; vt < 4; ++vt) {
      #pragma unroll
      for (int jt = 0; jt < 2; ++jt) {
        float rr = jt ? r1 : r0;
        #pragma unroll
        for (int i = 0; i < 8; ++i) {
          int vd = w * 64 + vt * 16 + lh * 8 + i;   // C/D layout: M = 8*lh + i
          int x  = q0 + jt * 16 + ln;               // N = ln
          size_t idx = (size_t)vd * HWDIM + x;
          float val = acc[vt][jt][i] * rr;
          if (br == 0) outp[idx] = val;
          else         outp[idx] += val;
        }
      }
    }
  }
}

// ---------------- launch ----------------

extern "C" void kernel_launch(void* const* d_in, const int* in_sizes, int n_in,
                              void* d_out, int out_size, void* d_ws, size_t ws_size,
                              hipStream_t stream) {
  const float* m_in   = (const float*)d_in[0];
  const float* m_out  = (const float*)d_in[1];
  const float* q_in   = (const float*)d_in[2];
  const float* q_out  = (const float*)d_in[3];
  const float* p_m_in = (const float*)d_in[4];
  const float* p_q_in = (const float*)d_in[5];
  float* out = (float*)d_out;

  char* ws = (char*)d_ws;
  unsigned short* m_in_h   = (unsigned short*)ws;                             // 4 MB
  unsigned short* p_m_in_h = (unsigned short*)(ws + (size_t)4 * 1024 * 1024); // 4 MB
  unsigned short* vt_h     = (unsigned short*)(ws + (size_t)8 * 1024 * 1024); // 8 MB

  const int nK = NO * CENTERS * CDIM;   // 2,097,152
  cvt_f16_kernel<<<dim3((nK / 2 + 255) / 256), dim3(256), 0, stream>>>(
      m_in, m_in_h, nK / 2);
  cvt_f16_kernel<<<dim3((nK / 2 + 255) / 256), dim3(256), 0, stream>>>(
      p_m_in, p_m_in_h, nK / 2);
  transpose_vt_kernel<<<dim3(CENTERS / 32, VDDIM / 32, NO), dim3(32, 8), 0, stream>>>(
      m_out, vt_h);

  const int nQ = NO * VDDIM * HWDIM;
  copy_qout_kernel<<<dim3((nQ + 255) / 256), dim3(256), 0, stream>>>(q_out, out);

  flash_attn_kernel<<<dim3(HWDIM / 32, NO), dim3(256), 0, stream>>>(
      m_in_h, p_m_in_h, q_in, p_q_in, vt_h, out);
}
